// TransformerModel_16655883174625
// MI455X (gfx1250) — matrix operand order repaired and checked by reading, not compiled, not run on hardware
//
#include <hip/hip_runtime.h>
#include <hip/hip_bf16.h>

// ---------------------------------------------------------------------------
// Model dims (fixed by the reference)
// ---------------------------------------------------------------------------
#define MB   4         // batch
#define MT   1024      // seq len
#define MD   768       // d_model
#define MH   12        // heads
#define MHD  64        // head dim
#define ML   6         // layers
#define MF   3072      // ffn dim
#define MV   32000     // vocab
#define MBT  (MB * MT) // tokens = 4096
#define LN_EPS 1e-5f

typedef __bf16 bf16_t;
typedef __attribute__((ext_vector_type(16))) __bf16 v16bf;
typedef __attribute__((ext_vector_type(8)))  __bf16 v8bf;
typedef __attribute__((ext_vector_type(8)))  float  v8f;
typedef __attribute__((ext_vector_type(2)))  float  v2f;

struct frag_bits { uint4 lo, hi; };
__device__ __forceinline__ v16bf load_frag(const unsigned* p) {
    frag_bits fb;
    fb.lo = *(const uint4*)(p);       // ds_load_b128
    fb.hi = *(const uint4*)(p + 4);   // ds_load_b128
    return __builtin_bit_cast(v16bf, fb);
}

// convert 8 fp32 -> 8 bf16 (RNE, native packed cvt) and store 16B to LDS
__device__ __forceinline__ void cvt_store8(unsigned* dst, const float* f) {
    v8f t;
    #pragma unroll
    for (int i = 0; i < 8; ++i) t[i] = f[i];
    v8bf b = __builtin_convertvector(t, v8bf);
    *(v8bf*)dst = b;                  // ds_store_b128
}

// ---------------------------------------------------------------------------
// GEMM: C[M,N] = op(A[M,K] @ B[K,N] + bias) (+ residual)
// Requirements (all call sites satisfy): M%128==0, N%128==0, K%32==0.
// Block tile 128x128, K-step 32, 256 threads = 8 waves.
// Wave w: rows 32*(w&3), cols 64*(w>>2) -> 2 A-frags x 4 B-frags = 8 WMMA/K-step.
// ---------------------------------------------------------------------------
__global__ __launch_bounds__(256)
void gemm_bf16_wmma(const float* __restrict__ A, const float* __restrict__ B,
                    const float* __restrict__ bias, const float* __restrict__ residual,
                    float* __restrict__ C, int M, int N, int K, int relu) {
    __shared__ __align__(16) unsigned AsU[8 * 32 * 12];  // 8 strips x 32 lanes x 24 bf16
    __shared__ __align__(16) unsigned BsU[8 * 32 * 12];  // 8 ntiles x 32 lanes x 24 bf16

    const int tid     = threadIdx.x;
    const int lane    = tid & 31;
    const int wave    = tid >> 5;
    const int wm      = wave & 3;
    const int wn      = wave >> 2;
    const int g       = lane >> 4;
    const int r       = lane & 15;
    const int rowBase = blockIdx.y * 128;
    const int colBase = blockIdx.x * 128;

    const int a_row   = tid >> 1;
    const int a_kc    = (tid & 1) * 16;
    const int a_strip = a_row >> 4;
    const int a_r     = a_row & 15;
    const int a_eo    = (a_kc ? 4 : 0);
    unsigned* a_dst0  = &AsU[((a_strip * 32 + a_kc + a_r) * 12)];
    unsigned* a_dst1  = &AsU[((a_strip * 32 + a_kc + a_r) * 12) + 4];
    const float* a_src = A + (size_t)(rowBase + a_row) * K + a_kc;

    const int b_n    = tid & 127;
    const int b_kc   = (tid >> 7) * 16;
    const int b_lane = (b_kc >> 4) * 16 + (b_n & 15);
    unsigned* b_dst  = &BsU[(((b_n >> 4) * 32 + b_lane) * 12)];
    const float* b_src = B + (size_t)b_kc * N + colBase + b_n;

    v8f c00 = {}, c01 = {}, c02 = {}, c03 = {};
    v8f c10 = {}, c11 = {}, c12 = {}, c13 = {};

    float ra[16];
    float rb[16];
    *(float4*)&ra[0]  = *(const float4*)(a_src);
    *(float4*)&ra[4]  = *(const float4*)(a_src + 4);
    *(float4*)&ra[8]  = *(const float4*)(a_src + 8);
    *(float4*)&ra[12] = *(const float4*)(a_src + 12);
    #pragma unroll
    for (int j = 0; j < 16; ++j) rb[j] = b_src[(size_t)j * N];

    const unsigned* a_frag0 = &AsU[(((wm * 2 + 0) * 32 + lane) * 12)];
    const unsigned* a_frag1 = &AsU[(((wm * 2 + 1) * 32 + lane) * 12)];
    const unsigned* b_frag0 = &BsU[(((wn * 4 + 0) * 32 + lane) * 12)];
    const unsigned* b_frag1 = &BsU[(((wn * 4 + 1) * 32 + lane) * 12)];
    const unsigned* b_frag2 = &BsU[(((wn * 4 + 2) * 32 + lane) * 12)];
    const unsigned* b_frag3 = &BsU[(((wn * 4 + 3) * 32 + lane) * 12)];

    for (int kb = 0; kb < K; kb += 32) {
        __syncthreads();
        cvt_store8(a_dst0, &ra[0]);
        cvt_store8(a_dst1, &ra[8]);
        cvt_store8(b_dst,     &rb[0]);
        cvt_store8(b_dst + 4, &rb[8]);
        __syncthreads();

        if (kb + 32 < K) {
            const float* an = a_src + kb + 32;
            *(float4*)&ra[0]  = *(const float4*)(an);
            *(float4*)&ra[4]  = *(const float4*)(an + 4);
            *(float4*)&ra[8]  = *(const float4*)(an + 8);
            *(float4*)&ra[12] = *(const float4*)(an + 12);
            const float* bn = b_src + (size_t)(kb + 32) * N;
            #pragma unroll
            for (int j = 0; j < 16; ++j) rb[j] = bn[(size_t)j * N];
        }

        v16bf af0 = load_frag(a_frag0);
        v16bf af1 = load_frag(a_frag1);
        v16bf bt;
        bt = load_frag(b_frag0);
        c00 = __builtin_amdgcn_wmma_f32_16x16x32_bf16(false, af0, false, bt, (short)0, c00, false, false);
        c10 = __builtin_amdgcn_wmma_f32_16x16x32_bf16(false, af1, false, bt, (short)0, c10, false, false);
        bt = load_frag(b_frag1);
        c01 = __builtin_amdgcn_wmma_f32_16x16x32_bf16(false, af0, false, bt, (short)0, c01, false, false);
        c11 = __builtin_amdgcn_wmma_f32_16x16x32_bf16(false, af1, false, bt, (short)0, c11, false, false);
        bt = load_frag(b_frag2);
        c02 = __builtin_amdgcn_wmma_f32_16x16x32_bf16(false, af0, false, bt, (short)0, c02, false, false);
        c12 = __builtin_amdgcn_wmma_f32_16x16x32_bf16(false, af1, false, bt, (short)0, c12, false, false);
        bt = load_frag(b_frag3);
        c03 = __builtin_amdgcn_wmma_f32_16x16x32_bf16(false, af0, false, bt, (short)0, c03, false, false);
        c13 = __builtin_amdgcn_wmma_f32_16x16x32_bf16(false, af1, false, bt, (short)0, c13, false, false);
    }

    const int colW = colBase + wn * 64 + r;
    #pragma unroll
    for (int sub = 0; sub < 2; ++sub) {
        #pragma unroll
        for (int v = 0; v < 8; ++v) {
            const int row = rowBase + wm * 32 + sub * 16 + g * 8 + v;
            float* crow = C + (size_t)row * N;
            const float* rrow = residual ? residual + (size_t)row * N : nullptr;
            #pragma unroll
            for (int nt = 0; nt < 4; ++nt) {
                const int col = colW + nt * 16;
                float val;
                if (sub == 0)
                    val = (nt == 0) ? c00[v] : (nt == 1) ? c01[v] : (nt == 2) ? c02[v] : c03[v];
                else
                    val = (nt == 0) ? c10[v] : (nt == 1) ? c11[v] : (nt == 2) ? c12[v] : c13[v];
                if (bias) val += bias[col];
                if (relu) val  = fmaxf(val, 0.0f);
                if (rrow) val += rrow[col];
                crow[col] = val;
            }
        }
    }
}

// ---------------------------------------------------------------------------
// LayerNorm over the last dim (cols). One block per row.
// ---------------------------------------------------------------------------
__global__ __launch_bounds__(256)
void layernorm_kernel(const float* __restrict__ x, const float* __restrict__ g,
                      const float* __restrict__ b, float* __restrict__ out, int cols) {
    __shared__ float red[256];
    const int row = blockIdx.x;
    const float* xr = x + (size_t)row * cols;
    float* orow     = out + (size_t)row * cols;

    float s = 0.0f, s2 = 0.0f;
    for (int c = threadIdx.x; c < cols; c += blockDim.x) {
        float f = xr[c];
        s += f; s2 += f * f;
    }
    red[threadIdx.x] = s;
    __syncthreads();
    for (int o = 128; o > 0; o >>= 1) {
        if (threadIdx.x < o) red[threadIdx.x] += red[threadIdx.x + o];
        __syncthreads();
    }
    float mean = red[0] / cols;
    __syncthreads();
    red[threadIdx.x] = s2;
    __syncthreads();
    for (int o = 128; o > 0; o >>= 1) {
        if (threadIdx.x < o) red[threadIdx.x] += red[threadIdx.x + o];
        __syncthreads();
    }
    float var  = red[0] / cols - mean * mean;
    float rstd = rsqrtf(var + LN_EPS);
    for (int c = threadIdx.x; c < cols; c += blockDim.x)
        orow[c] = (xr[c] - mean) * rstd * g[c] + b[c];
}

// ---------------------------------------------------------------------------
// Flash attention on the matrix pipe: V_WMMA_F32_16X16X4_F32 (true fp32 WMMA).
// One wave per 16-query tile of one (b,h). Q,O: [B,T,D]; K,V: [B,S,D].
// Score rows (queries) live on (vgpr v, lane-group g); cols (keys) on lanes r.
// ---------------------------------------------------------------------------
__global__ __launch_bounds__(32)
void attention_wmma_kernel(const float* __restrict__ Q, const float* __restrict__ Km,
                           const float* __restrict__ Vm, float* __restrict__ O,
                           int S, int causal) {
    __shared__ __align__(8) float Pbuf[16 * 16]; // P transpose staging

    const int qt = blockIdx.x, h = blockIdx.y, b = blockIdx.z;
    const int lane = threadIdx.x & 31;
    const int g = lane >> 4;
    const int r = lane & 15;
    const int hd0 = h * MHD;
    const float scale = 0.125f; // 1/sqrt(64)

    const float* Qb = Q  + ((size_t)b * MT + (size_t)qt * 16) * MD + hd0;
    const float* Kb = Km + (size_t)b * S * MD + hd0;
    const float* Vb = Vm + (size_t)b * S * MD + hd0;

    // Q fragments (A of 16x16x4): lane holds Q[r][4j+2g .. +1], scale folded in
    v2f qf[16];
    #pragma unroll
    for (int j = 0; j < 16; ++j) {
        const float* p = Qb + (size_t)r * MD + 4 * j + 2 * g;
        qf[j][0] = p[0] * scale;
        qf[j][1] = p[1] * scale;
    }

    v8f o0 = {}, o1 = {}, o2 = {}, o3 = {};
    float mrow[8], lrow[8];
    #pragma unroll
    for (int v = 0; v < 8; ++v) { mrow[v] = -1e30f; lrow[v] = 0.0f; }

    const int nTiles = causal ? (qt + 1) : (S >> 4);
    for (int kb = 0; kb < nTiles; ++kb) {
        // ---- scores: 16 chained fp32 WMMAs, C = Q @ K^T (16x16)
        v8f c = {};
        #pragma unroll
        for (int j = 0; j < 16; ++j) {
            const float* kp = Kb + (size_t)(kb * 16 + r) * MD + 4 * j + 2 * g;
            v2f bfr;
            bfr[0] = kp[0];
            bfr[1] = kp[1];
            c = __builtin_amdgcn_wmma_f32_16x16x4_f32(false, qf[j], false, bfr,
                                                      (short)0, c, false, false);
        }
        // causal mask on the diagonal tile (key s = kb*16+r, query t = qt*16+v+8g)
        if (causal && kb == qt) {
            #pragma unroll
            for (int v = 0; v < 8; ++v)
                if (r > v + 8 * g) c[v] = -1e30f;
        }
        // ---- online softmax; row stats reduced across 16 lanes of each group
        float alpha[8];
        #pragma unroll
        for (int v = 0; v < 8; ++v) {
            float mx = c[v];
            mx = fmaxf(mx, __shfl_xor(mx, 1, 16));
            mx = fmaxf(mx, __shfl_xor(mx, 2, 16));
            mx = fmaxf(mx, __shfl_xor(mx, 4, 16));
            mx = fmaxf(mx, __shfl_xor(mx, 8, 16));
            float mnew = fmaxf(mrow[v], mx);
            alpha[v] = __expf(mrow[v] - mnew);
            mrow[v] = mnew;
            float p = __expf(c[v] - mnew);
            c[v] = p;
            float sum = p;
            sum += __shfl_xor(sum, 1, 16);
            sum += __shfl_xor(sum, 2, 16);
            sum += __shfl_xor(sum, 4, 16);
            sum += __shfl_xor(sum, 8, 16);
            lrow[v] = lrow[v] * alpha[v] + sum;
        }
        #pragma unroll
        for (int v = 0; v < 8; ++v) {
            o0[v] *= alpha[v]; o1[v] *= alpha[v];
            o2[v] *= alpha[v]; o3[v] *= alpha[v];
        }
        // ---- transpose P through LDS (C layout -> A-fragment layout)
        #pragma unroll
        for (int v = 0; v < 8; ++v)
            Pbuf[(v + 8 * g) * 16 + r] = c[v];
        __syncthreads();   // single-wave workgroup: cheap
        v2f pa[4];
        #pragma unroll
        for (int j = 0; j < 4; ++j) {
            pa[j][0] = Pbuf[r * 16 + 4 * j + 2 * g];
            pa[j][1] = Pbuf[r * 16 + 4 * j + 2 * g + 1];
        }
        __syncthreads();
        // ---- PV accumulate: 16 chained fp32 WMMAs (4 K-chunks x 4 col tiles)
        #pragma unroll
        for (int j = 0; j < 4; ++j) {
            const float* vr0 = Vb + (size_t)(kb * 16 + 4 * j + 2 * g) * MD;
            const float* vr1 = vr0 + MD;
            v2f bv;
            bv[0] = vr0[0 * 16 + r]; bv[1] = vr1[0 * 16 + r];
            o0 = __builtin_amdgcn_wmma_f32_16x16x4_f32(false, pa[j], false, bv, (short)0, o0, false, false);
            bv[0] = vr0[1 * 16 + r]; bv[1] = vr1[1 * 16 + r];
            o1 = __builtin_amdgcn_wmma_f32_16x16x4_f32(false, pa[j], false, bv, (short)0, o1, false, false);
            bv[0] = vr0[2 * 16 + r]; bv[1] = vr1[2 * 16 + r];
            o2 = __builtin_amdgcn_wmma_f32_16x16x4_f32(false, pa[j], false, bv, (short)0, o2, false, false);
            bv[0] = vr0[3 * 16 + r]; bv[1] = vr1[3 * 16 + r];
            o3 = __builtin_amdgcn_wmma_f32_16x16x4_f32(false, pa[j], false, bv, (short)0, o3, false, false);
        }
    }

    // ---- final normalize + store: O[qt*16 + v + 8g][hd0 + nt*16 + r]
    #pragma unroll
    for (int v = 0; v < 8; ++v) {
        float invl = 1.0f / lrow[v];
        float* op = O + ((size_t)b * MT + (size_t)qt * 16 + v + 8 * g) * MD + hd0;
        op[0 * 16 + r] = o0[v] * invl;
        op[1 * 16 + r] = o1[v] * invl;
        op[2 * 16 + r] = o2[v] * invl;
        op[3 * 16 + r] = o3[v] * invl;
    }
}

// ---------------------------------------------------------------------------
// Embedding: out[b,t,:] = tok_emb[idx[b,t],:] + pos_emb[t,:]
// ---------------------------------------------------------------------------
__global__ __launch_bounds__(256)
void embed_kernel(const int* __restrict__ idx, const float* __restrict__ tok,
                  const float* __restrict__ pos, float* __restrict__ out) {
    const int bt = blockIdx.x;
    const int t  = bt % MT;
    const int id = idx[bt];
    const float* tp = tok + (size_t)id * MD;
    const float* pp = pos + (size_t)t * MD;
    float* op = out + (size_t)bt * MD;
    for (int d = threadIdx.x; d < MD; d += blockDim.x)
        op[d] = tp[d] + pp[d];
}

// ---------------------------------------------------------------------------
// Host-side orchestration
// ---------------------------------------------------------------------------
static inline void run_gemm(const float* A, const float* Bm, const float* bias,
                            const float* res, float* C, int M, int N, int K,
                            int relu, hipStream_t s) {
    dim3 grid(N / 128, M / 128);
    gemm_bf16_wmma<<<grid, 256, 0, s>>>(A, Bm, bias, res, C, M, N, K, relu);
}
static inline void run_ln(const float* x, const float* g, const float* b,
                          float* out, hipStream_t s) {
    layernorm_kernel<<<MBT, 256, 0, s>>>(x, g, b, out, MD);
}
static inline void run_attn(const float* Q, const float* K, const float* V,
                            float* O, int S, int causal, hipStream_t s) {
    dim3 grid(MT / 16, MH, MB);
    attention_wmma_kernel<<<grid, 32, 0, s>>>(Q, K, V, O, S, causal);
}

extern "C" void kernel_launch(void* const* d_in, const int* in_sizes, int n_in,
                              void* d_out, int out_size, void* d_ws, size_t ws_size,
                              hipStream_t stream) {
    // ---- inputs (setup_inputs() order) ----
    const int*   src_idx     = (const int*)  d_in[0];
    const int*   tgt_idx     = (const int*)  d_in[1];
    const float* enc_tok_emb = (const float*)d_in[2];
    const float* enc_pos_emb = (const float*)d_in[3];
    const float* enc_Wq      = (const float*)d_in[4];
    const float* enc_Wk      = (const float*)d_in[5];
    const float* enc_Wv      = (const float*)d_in[6];
    const float* enc_Wo      = (const float*)d_in[7];
    const float* enc_bo      = (const float*)d_in[8];
    const float* enc_ln1_g   = (const float*)d_in[9];
    const float* enc_ln1_b   = (const float*)d_in[10];
    const float* enc_W1      = (const float*)d_in[11];
    const float* enc_b1      = (const float*)d_in[12];
    const float* enc_W2      = (const float*)d_in[13];
    const float* enc_b2      = (const float*)d_in[14];
    const float* enc_ln2_g   = (const float*)d_in[15];
    const float* enc_ln2_b   = (const float*)d_in[16];
    const float* enc_lnf_g   = (const float*)d_in[17];
    const float* enc_lnf_b   = (const float*)d_in[18];
    const float* dec_tok_emb = (const float*)d_in[19];
    const float* dec_pos_emb = (const float*)d_in[20];
    const float* dec_sWq     = (const float*)d_in[21];
    const float* dec_sWk     = (const float*)d_in[22];
    const float* dec_sWv     = (const float*)d_in[23];
    const float* dec_sWo     = (const float*)d_in[24];
    const float* dec_sbo     = (const float*)d_in[25];
    const float* dec_cWq     = (const float*)d_in[26];
    const float* dec_cWk     = (const float*)d_in[27];
    const float* dec_cWv     = (const float*)d_in[28];
    const float* dec_cWo     = (const float*)d_in[29];
    const float* dec_cbo     = (const float*)d_in[30];
    const float* dec_ln1_g   = (const float*)d_in[31];
    const float* dec_ln1_b   = (const float*)d_in[32];
    const float* dec_ln2_g   = (const float*)d_in[33];
    const float* dec_ln2_b   = (const float*)d_in[34];
    const float* dec_ln3_g   = (const float*)d_in[35];
    const float* dec_ln3_b   = (const float*)d_in[36];
    const float* dec_W1      = (const float*)d_in[37];
    const float* dec_b1      = (const float*)d_in[38];
    const float* dec_W2      = (const float*)d_in[39];
    const float* dec_b2      = (const float*)d_in[40];
    const float* dec_lnf_g   = (const float*)d_in[41];
    const float* dec_lnf_b   = (const float*)d_in[42];
    const float* lm_W        = (const float*)d_in[43];
    const float* lm_b        = (const float*)d_in[44];
    float* logits = (float*)d_out;
    (void)in_sizes; (void)n_in; (void)out_size; (void)ws_size;

    // ---- workspace layout (fp32) ----
    const size_t ND = (size_t)MBT * MD;
    float* ws  = (float*)d_ws;
    float* x   = ws;            // encoder stream
    float* h   = x   + ND;      // LN output
    float* q   = h   + ND;
    float* k   = q   + ND;
    float* v   = k   + ND;
    float* ao  = v   + ND;      // attention output (pre-proj)
    float* enc = ao  + ND;      // encoder final
    float* y   = enc + ND;      // decoder stream
    float* ff  = y   + ND;      // FFN hidden [MBT, MF]

    const size_t DD = (size_t)MD * MD;
    const size_t DF = (size_t)MD * MF;

    // ================= Encoder =================
    embed_kernel<<<MBT, 256, 0, stream>>>(src_idx, enc_tok_emb, enc_pos_emb, x);
    for (int l = 0; l < ML; ++l) {
        run_ln(x, enc_ln1_g + l * MD, enc_ln1_b + l * MD, h, stream);
        run_gemm(h, enc_Wq + l * DD, nullptr, nullptr, q, MBT, MD, MD, 0, stream);
        run_gemm(h, enc_Wk + l * DD, nullptr, nullptr, k, MBT, MD, MD, 0, stream);
        run_gemm(h, enc_Wv + l * DD, nullptr, nullptr, v, MBT, MD, MD, 0, stream);
        run_attn(q, k, v, ao, MT, /*causal=*/0, stream);
        run_gemm(ao, enc_Wo + l * DD, enc_bo + l * MD, x, x, MBT, MD, MD, 0, stream);
        run_ln(x, enc_ln2_g + l * MD, enc_ln2_b + l * MD, h, stream);
        run_gemm(h, enc_W1 + l * DF, enc_b1 + l * MF, nullptr, ff, MBT, MF, MD, 1, stream);
        run_gemm(ff, enc_W2 + l * DF, enc_b2 + l * MD, x, x, MBT, MD, MF, 0, stream);
    }
    run_ln(x, enc_lnf_g, enc_lnf_b, enc, stream);

    // ================= Decoder =================
    embed_kernel<<<MBT, 256, 0, stream>>>(tgt_idx, dec_tok_emb, dec_pos_emb, y);
    for (int l = 0; l < ML; ++l) {
        // self-attention: Q from ln1(y), K/V from pre-LN y (reference quirk)
        run_ln(y, dec_ln1_g + l * MD, dec_ln1_b + l * MD, h, stream);
        run_gemm(h, dec_sWq + l * DD, nullptr, nullptr, q, MBT, MD, MD, 0, stream);
        run_gemm(y, dec_sWk + l * DD, nullptr, nullptr, k, MBT, MD, MD, 0, stream);
        run_gemm(y, dec_sWv + l * DD, nullptr, nullptr, v, MBT, MD, MD, 0, stream);
        run_attn(q, k, v, ao, MT, /*causal=*/1, stream);
        run_gemm(ao, dec_sWo + l * DD, dec_sbo + l * MD, y, y, MBT, MD, MD, 0, stream);
        // cross-attention: K/V from encoder output
        run_ln(y, dec_ln2_g + l * MD, dec_ln2_b + l * MD, h, stream);
        run_gemm(h,   dec_cWq + l * DD, nullptr, nullptr, q, MBT, MD, MD, 0, stream);
        run_gemm(enc, dec_cWk + l * DD, nullptr, nullptr, k, MBT, MD, MD, 0, stream);
        run_gemm(enc, dec_cWv + l * DD, nullptr, nullptr, v, MBT, MD, MD, 0, stream);
        run_attn(q, k, v, ao, MT, /*causal=*/0, stream);
        run_gemm(ao, dec_cWo + l * DD, dec_cbo + l * MD, y, y, MBT, MD, MD, 0, stream);
        // FFN
        run_ln(y, dec_ln3_g + l * MD, dec_ln3_b + l * MD, h, stream);
        run_gemm(h, dec_W1 + l * DF, dec_b1 + l * MF, nullptr, ff, MBT, MF, MD, 1, stream);
        run_gemm(ff, dec_W2 + l * DF, dec_b2 + l * MD, y, y, MBT, MD, MF, 0, stream);
    }
    run_ln(y, dec_lnf_g, dec_lnf_b, h, stream);

    // ================= LM head =================
    run_gemm(h, lm_W, lm_b, nullptr, logits, MBT, MV, MD, 0, stream);
}
